// TemporalMaskModelAttention_87582973100391
// MI455X (gfx1250) — matrix operand order repaired and checked by reading, not compiled, not run on hardware
//
#include <hip/hip_runtime.h>

// ---------------------------------------------------------------------------
// TemporalMaskModelAttention on MI455X (gfx1250)
// 3x3 convs as 9-shift implicit GEMMs on v_wmma_f32_16x16x32_bf16.
// Weight slabs staged via TDM (tensor_load_to_lds, TENSORcnt),
// bias via global_load_async_to_lds (ASYNCcnt), halo tile via LDS transpose.
// ---------------------------------------------------------------------------

typedef __attribute__((ext_vector_type(16))) __bf16         v16bf;
typedef __attribute__((ext_vector_type(8)))  float          v8f;
typedef __attribute__((ext_vector_type(8)))  unsigned short us8;
typedef __attribute__((ext_vector_type(4)))  unsigned int   v4u;
typedef __attribute__((ext_vector_type(8)))  unsigned int   v8u;

static constexpr int Hh = 64, Ww = 192, P = Hh * Ww;     // 12288 pixels
static constexpr int PH = 66, PW = 208, PLANE = PH * PW; // padded plane (13728)

__device__ __forceinline__ unsigned short f2bf(float f) {
  unsigned u = __float_as_uint(f);
  u += 0x7fffu + ((u >> 16) & 1u);          // round-to-nearest-even
  return (unsigned short)(u >> 16);
}

union FragAB { us8 h[2]; v16bf v; };

// ---------------- f32 -> padded bf16 plane ----------------
__global__ void cvt_pad_bf16(const float* __restrict__ s,
                             unsigned short* __restrict__ d, long planes) {
  long n  = planes * P;
  long i  = (long)blockIdx.x * blockDim.x + threadIdx.x;
  long st = (long)gridDim.x * blockDim.x;
  for (; i < n; i += st) {
    long pl = i / P;
    int  p  = (int)(i - pl * P);
    int  yy = p / Ww, xx = p - yy * Ww;
    d[pl * PLANE + (size_t)(yy + 1) * PW + xx + 8] = f2bf(s[i]);
  }
}

// ------- weight repack: (O,I,3,3) f32 -> [shift][R][I] bf16 (R = O rounded to 64)
__global__ void cvt_wshift(const float* __restrict__ s,
                           unsigned short* __restrict__ d, int Cout, int Cin, int R) {
  int n  = 9 * Cout * Cin;
  int st = gridDim.x * blockDim.x;
  for (int j = blockIdx.x * blockDim.x + threadIdx.x; j < n; j += st) {
    int sft = j / (Cout * Cin);
    int r   = j - sft * (Cout * Cin);
    int o   = r / Cin;
    int c   = r - o * Cin;
    d[(size_t)(sft * R + o) * Cin + c] = f2bf(s[(o * Cin + c) * 9 + sft]);
  }
}

// ---------------- 3x3 SAME conv: 9-shift implicit GEMM, bf16 WMMA -----------
// inP : [Nmaps][Cin][PH][PW] padded bf16 (zero border)
// wS  : [9][R][Cin] bf16 (shift-major, rows padded to R = roundup64(Cout))
// out : [Nmaps][Cout][P] f32
// block tile: 64 Cout x 64 pixels, 128 threads (4 waves; 1 M-tile x 4 N-tiles each)
__global__ void conv3x3_wmma(const unsigned short* __restrict__ inP,
                             const unsigned short* __restrict__ wS,
                             const float* __restrict__ bias,
                             float* __restrict__ out, int Cin, int Cout) {
  const int NT   = P / 64;                  // 192 tiles; 64 px never cross a row
  const int nimg = blockIdx.x / NT;
  const int t64  = blockIdx.x % NT;
  const int p0   = t64 * 64;
  const int y    = p0 / Ww;
  const int x0   = p0 % Ww;
  const int t    = threadIdx.x;
  const int lane = t & 31;
  const int wave = t >> 5;
  const int oBase = blockIdx.y * 64;
  const int R     = (Cout + 63) & ~63;      // padded weight rows

  // BL: input halo tile, transposed: [(ry*80 + rx)*40 + ch], 19200 B
  // WL: TDM-staged weights:          [(s*64 + o)*32 + c],    36864 B
  __shared__ __align__(16) unsigned short BL[3 * 80 * 40];
  __shared__ __align__(16) unsigned short WL[9 * 64 * 32];
  __shared__ __align__(16) float          BIAS[64];

  const int kb  = (lane < 16) ? 0 : 8;    // A-fragment K half
  const int kbB = (lane < 16) ? 0 : 8;    // B-fragment K half
  const int ml  = wave * 16 + (lane & 15);

  // ---- stage bias through the async global->LDS path (ASYNCcnt) ----
  const bool biasLds = (oBase + 64 <= Cout);
  if (biasLds && t < 16) {
    unsigned ldst = (unsigned)(uintptr_t)&BIAS[t * 4];
    unsigned voff = (unsigned)((oBase + t * 4) * 4);
    asm volatile("global_load_async_to_lds_b128 %0, %1, %2"
                 :: "v"(ldst), "v"(voff),
                    "s"((unsigned long long)(uintptr_t)bias) : "memory");
  }
  asm volatile("s_wait_asynccnt 0x0" ::: "memory");

  v8f acc[4] = {};

  for (int c0 = 0; c0 < Cin; c0 += 32) {
    __syncthreads();   // WAR: previous iteration's LDS reads done

    // ---- TDM: one tensor_load_to_lds for the [9][64][32] weight slab ----
    if (wave == 0) {
      unsigned long long ga = (unsigned long long)(uintptr_t)wS
                            + (((unsigned long long)oBase * Cin + c0) * 2ull);
      const unsigned td0 = (unsigned)Cin;          // tensor dim0 (x: channels)
      const unsigned td1 = (unsigned)R;            // tensor dim1 (y: out rows)
      const unsigned zst = (unsigned)(R * Cin);    // dim1 (z) stride, elements
      v4u g0;
      g0.x = 1u;                                   // count=1, user descriptor
      g0.y = (unsigned)(uintptr_t)&WL[0];          // lds_addr (bytes)
      g0.z = (unsigned)ga;                         // global_addr[31:0]
      g0.w = (unsigned)((ga >> 32) & 0x01ffffffull) | (2u << 30);  // [56:32]|type=2
      v8u g1;
      g1.s0 = 1u << 16;                            // data_size = 2 bytes
      g1.s1 = (td0 & 0xffffu) << 16;               // tensor_dim0[15:0]
      g1.s2 = (td0 >> 16) | ((td1 & 0xffffu) << 16);   // td0[31:16] | td1[15:0]
      g1.s3 = (td1 >> 16) | (32u << 16);           // td1[31:16] | tile_dim0=32
      g1.s4 = 64u | (9u << 16);                    // tile_dim1=64 | tile_dim2=9
      g1.s5 = td0;                                 // tensor_dim0_stride[31:0]=Cin
      g1.s6 = (zst & 0xffffu) << 16;               // strd0[47:32]=0 | strd1[15:0]
      g1.s7 = zst >> 16;                           // tensor_dim1_stride[47:16]
      v4u g2;
      g2.x = 9u;                                   // tensor_dim2
      g2.y = 0u; g2.z = 0u; g2.w = 0u;             // no dim3 / tile_dim3
      asm volatile("tensor_load_to_lds %0, %1, %2, %2"
                   :: "s"(g0), "s"(g1), "s"(g2) : "memory");
      __builtin_amdgcn_s_wait_tensorcnt(0);
    }

    // ---- stage input halo tile: 32 ch x 3 rows x 80 px, transpose into BL
    for (int i = t; i < 960; i += 128) {
      int ch = i / 30, r2 = i - ch * 30, ry = r2 / 10, rq = r2 - ry * 10;
      const unsigned short* gp = inP +
          ((size_t)(nimg * Cin + c0 + ch) * PH + y + ry) * PW + x0 + rq * 8;
      us8 dv = *(const us8*)gp;
      #pragma unroll
      for (int j = 0; j < 8; ++j)
        BL[(ry * 80 + rq * 8 + j) * 40 + ch] = dv[j];
    }
    if (c0 + 32 < Cin)
      __builtin_prefetch(inP + ((size_t)(nimg * Cin + c0 + 32) * PH + y) * PW + x0, 0, 2);

    __syncthreads();   // TDM copy + halo transpose visible to all waves

    // ---- software-pipelined 9 shifts x 4 N-tiles of WMMA ----
    auto loadA = [&](int s) -> FragAB {
      const unsigned short* ap = &WL[(s * 64 + ml) * 32 + kb];
      FragAB f; f.h[0] = *(const us8*)ap; f.h[1] = *(const us8*)(ap + 16); return f;
    };
    auto loadB = [&](int s, int nt) -> FragAB {
      const int dy = s / 3, dx = s % 3;
      const int px = nt * 16 + (lane & 15);
      const unsigned short* bp = &BL[(dy * 80 + px + dx + 7) * 40 + kbB];
      FragAB f; f.h[0] = *(const us8*)bp; f.h[1] = *(const us8*)(bp + 16); return f;
    };
    FragAB a = loadA(0);
    FragAB b = loadB(0, 0);
    #pragma unroll
    for (int it = 0; it < 36; ++it) {
      FragAB an = a, bn = b;
      if (it < 35) {                       // one-step lookahead hides ds latency
        const int j = it + 1;
        bn = loadB(j >> 2, j & 3);
        if ((j & 3) == 0) an = loadA(j >> 2);
      }
      acc[it & 3] = __builtin_amdgcn_wmma_f32_16x16x32_bf16(false, a.v, false, b.v,
                                                            (short)0, acc[it & 3],
                                                            false, false);
      a = an; b = bn;
    }
  }

  const int rowAdd = (lane < 16) ? 0 : 8;
  #pragma unroll
  for (int nt = 0; nt < 4; ++nt) {
    const size_t col = (size_t)p0 + nt * 16 + (lane & 15);
    #pragma unroll
    for (int r = 0; r < 8; ++r) {
      int row = oBase + wave * 16 + r + rowAdd;
      if (row < Cout) {
        float bv = biasLds ? BIAS[row - oBase] : bias[row];
        out[((size_t)nimg * Cout + row) * P + col] = acc[nt][r] + bv;
      }
    }
  }
}

// ---------------- q.k channel dots ----------------
__global__ void dot_qk(const float* __restrict__ q, const float* __restrict__ k,
                       float* __restrict__ s) {
  const int mc = blockIdx.x;
  const float* qp = q + (size_t)mc * P;
  const float* kp = k + (size_t)mc * P;
  float a = 0.f;
  for (int p = threadIdx.x; p < P; p += 256) a += qp[p] * kp[p];
  __shared__ float sh[256];
  sh[threadIdx.x] = a; __syncthreads();
  for (int o = 128; o; o >>= 1) {
    if (threadIdx.x < o) sh[threadIdx.x] += sh[threadIdx.x + o];
    __syncthreads();
  }
  if (!threadIdx.x) s[mc] = sh[0];
}

// ---------------- softmax over 64 channels, one block per map ----------------
__global__ void softmax64(const float* __restrict__ s, float* __restrict__ sm) {
  const int t = threadIdx.x;
  const float v = s[blockIdx.x * 64 + t];
  __shared__ float sh[64];
  sh[t] = v; __syncthreads();
  for (int o = 32; o; o >>= 1) { if (t < o) sh[t] = fmaxf(sh[t], sh[t + o]); __syncthreads(); }
  const float mx = sh[0]; __syncthreads();
  const float e = expf(v - mx);
  sh[t] = e; __syncthreads();
  for (int o = 32; o; o >>= 1) { if (t < o) sh[t] += sh[t + o]; __syncthreads(); }
  sm[blockIdx.x * 64 + t] = e / sh[0];
}

// ------- softmax-weighted aggregation (mean over member maps) -> padded bf16
__global__ void ctx_agg_k(const float* __restrict__ vin, const float* __restrict__ sm,
                          unsigned short* __restrict__ outP,
                          int start, int stride, int count) {
  const int c = blockIdx.x;
  const int p = blockIdx.y * blockDim.x + threadIdx.x;
  float a = 0.f;
  for (int m = 0; m < count; ++m) {
    int map = start + m * stride;
    a += sm[map * 64 + c] * vin[((size_t)map * 64 + c) * P + p];
  }
  const int yy = p / Ww, xx = p - (p / Ww) * Ww;
  outP[(size_t)c * PLANE + (size_t)(yy + 1) * PW + xx + 8] = f2bf(a / (float)count);
}

// ---------------- batch-norm statistics: one block per (group,channel) ------
__global__ void bn_stats_k(const float* __restrict__ x, float* __restrict__ mean,
                           float* __restrict__ inv, int B, int C) {
  const int gc = blockIdx.x;
  const int g  = gc / C, c = gc % C;
  __shared__ float s1[256], s2[256];
  float a = 0.f, q = 0.f;
  for (int bb = 0; bb < B; ++bb) {
    const float* xp = x + (((size_t)(g * B + bb)) * C + c) * P;
    for (int p = threadIdx.x; p < P; p += 256) { float t = xp[p]; a += t; q += t * t; }
  }
  s1[threadIdx.x] = a; s2[threadIdx.x] = q; __syncthreads();
  for (int o = 128; o; o >>= 1) {
    if (threadIdx.x < o) { s1[threadIdx.x] += s1[threadIdx.x + o]; s2[threadIdx.x] += s2[threadIdx.x + o]; }
    __syncthreads();
  }
  if (!threadIdx.x) {
    float n = (float)B * (float)P;
    float m = s1[0] / n;
    float var = s2[0] / n - m * m;
    mean[gc] = m;
    inv[gc]  = rsqrtf(var + 1e-5f);
  }
}

// ---------------- BN + ReLU -> padded bf16 ----------------
__global__ void bn_relu_bf16_k(const float* __restrict__ x, const float* __restrict__ gma,
                               const float* __restrict__ bta, const float* __restrict__ mean,
                               const float* __restrict__ inv, unsigned short* __restrict__ outP,
                               int B, int C) {
  const size_t i = blockIdx.x;                 // (g*B + b)*C + c
  const int c = (int)(i % C);
  const int grp = (int)(i / C) / B;
  const int gc = grp * C + c;
  const int p = blockIdx.y * blockDim.x + threadIdx.x;
  float val = x[i * P + p];
  val = gma[c] * (val - mean[gc]) * inv[gc] + bta[c];
  const int yy = p / Ww, xx = p - (p / Ww) * Ww;
  outP[i * PLANE + (size_t)(yy + 1) * PW + xx + 8] = f2bf(fmaxf(val, 0.f));
}

// ---------------- BN + sigmoid -> f32 (mask head) ----------------
__global__ void bn_sigmoid_f32_k(const float* __restrict__ x, const float* __restrict__ gma,
                                 const float* __restrict__ bta, const float* __restrict__ mean,
                                 const float* __restrict__ inv, float* __restrict__ out,
                                 int B, int C) {
  const size_t i = blockIdx.x;
  const int c = (int)(i % C);
  const int grp = (int)(i / C) / B;
  const int gc = grp * C + c;
  const int p = blockIdx.y * blockDim.x + threadIdx.x;
  float val = x[i * P + p];
  val = gma[c] * (val - mean[gc]) * inv[gc] + bta[c];
  out[i * P + p] = 1.f / (1.f + expf(-val));
}

// ---------------- BN + ReLU, multiplied by mask -> f32 output ----------------
__global__ void bn_relu_mask_f32_k(const float* __restrict__ x, const float* __restrict__ gma,
                                   const float* __restrict__ bta, const float* __restrict__ mean,
                                   const float* __restrict__ inv, const float* __restrict__ mask,
                                   float* __restrict__ out, int B, int C) {
  const size_t i = blockIdx.x;                 // b*C + c  (G == 1)
  const int c = (int)(i % C);
  const int b = (int)(i / C) % B;
  const int p = blockIdx.y * blockDim.x + threadIdx.x;
  float val = x[i * P + p];
  val = gma[c] * (val - mean[c]) * inv[c] + bta[c];
  out[i * P + p] = fmaxf(val, 0.f) * mask[(size_t)b * P + p];
}

// ---------------------------------------------------------------------------
extern "C" void kernel_launch(void* const* d_in, const int* in_sizes, int n_in,
                              void* d_out, int out_size, void* d_ws, size_t ws_size,
                              hipStream_t stream) {
  (void)in_sizes; (void)n_in; (void)out_size; (void)ws_size;

  const float* feat = (const float*)d_in[0];
  const float* cf_qw  = (const float*)d_in[2];
  const float* cf_qb  = (const float*)d_in[3];
  const float* cf_kw  = (const float*)d_in[4];
  const float* cf_kb  = (const float*)d_in[5];
  const float* cf_vw  = (const float*)d_in[6];
  const float* cf_vb  = (const float*)d_in[7];
  const float* cf_o1w = (const float*)d_in[8];
  const float* cf_o1b = (const float*)d_in[9];
  const float* cf_g1  = (const float*)d_in[10];
  const float* cf_b1  = (const float*)d_in[11];
  const float* cf_o2w = (const float*)d_in[12];
  const float* cf_o2b = (const float*)d_in[13];
  const float* cf_g2  = (const float*)d_in[14];
  const float* cf_b2  = (const float*)d_in[15];
  const float* tf_qw  = (const float*)d_in[16];
  const float* tf_qb  = (const float*)d_in[17];
  const float* tf_kw  = (const float*)d_in[18];
  const float* tf_kb  = (const float*)d_in[19];
  const float* tf_vw  = (const float*)d_in[20];
  const float* tf_vb  = (const float*)d_in[21];
  const float* tf_o1w = (const float*)d_in[22];
  const float* tf_o1b = (const float*)d_in[23];
  const float* tf_g1  = (const float*)d_in[24];
  const float* tf_b1  = (const float*)d_in[25];
  const float* tf_o2w = (const float*)d_in[26];
  const float* tf_o2b = (const float*)d_in[27];
  const float* tf_g2  = (const float*)d_in[28];
  const float* tf_b2  = (const float*)d_in[29];
  const float* mm_w1  = (const float*)d_in[30];
  const float* mm_b1  = (const float*)d_in[31];
  const float* mm_g1  = (const float*)d_in[32];
  const float* mm_be1 = (const float*)d_in[33];
  const float* mm_w2  = (const float*)d_in[34];
  const float* mm_b2  = (const float*)d_in[35];
  const float* mm_g2  = (const float*)d_in[36];
  const float* mm_be2 = (const float*)d_in[37];
  const float* ol_w   = (const float*)d_in[38];
  const float* ol_b   = (const float*)d_in[39];
  const float* ol_g   = (const float*)d_in[40];
  const float* ol_be  = (const float*)d_in[41];

  char* WS = (char*)d_ws;
  const size_t MB = 1ull << 20;

  // ---- workspace layout (phase-based aliasing) ----
  unsigned short* FEATP  = (unsigned short*)(WS);             // 24*256 padded planes
  unsigned short* WBF    = (unsigned short*)(WS + 176 * MB);  // shift-major weights
  float* Q  = (float*)(WS + 184 * MB);                        // 24*64*P f32
  float* Kk = (float*)(WS + 256 * MB);
  float* V  = (float*)(WS + 328 * MB);
  float* ST = (float*)(WS + 400 * MB);
  unsigned short* AGGP   = (unsigned short*)(WS + 402 * MB);  // 8*64 padded
  unsigned short* Y1P    = (unsigned short*)(WS + 418 * MB);  // 8*256 padded
  unsigned short* FEATSP = (unsigned short*)(WS + 476 * MB);  // 8*256 padded
  unsigned short* AGGTP  = (unsigned short*)(WS + 534 * MB);  // 2*64 padded
  unsigned short* YTP    = (unsigned short*)(WS + 538 * MB);  // 2*256 padded
  unsigned short* HFP    = (unsigned short*)(WS + 554 * MB);  // 2*256 padded
  unsigned short* HBP    = (unsigned short*)(WS + 570 * MB);  // 2*64 padded
  // f32 conv outputs aliased into FEATP region once cf q/k/v convs are done
  float* O1  = (float*)(WS);                 // 8*256*P (96MB)
  float* O2  = (float*)(WS);                 // reuses O1 (dead by then)
  float* QT  = (float*)(WS);                 // 6*64*P
  float* KT  = (float*)(WS + 20 * MB);
  float* VT  = (float*)(WS + 40 * MB);
  float* O1T = (float*)(WS + 60 * MB);       // 2*256*P (24MB)
  float* O2T = (float*)(WS + 96 * MB);
  float* MM1 = (float*)(WS + 128 * MB);      // 2*64*P
  float* MM2 = (float*)(WS + 136 * MB);      // 2*1*P
  float* OLO = (float*)(WS + 140 * MB);      // 2*256*P

  float* S24  = ST;
  float* SM24 = ST + 2048;
  float* MEAN = ST + 4096;
  float* INV  = ST + 6144;
  float* S6   = ST + 8192;
  float* SM6  = ST + 8704;

  // ---- zero padded activation buffers + weight region (pad rows must be 0) ----
  hipMemsetAsync(FEATP,  0, (size_t)24 * 256 * PLANE * 2, stream);
  hipMemsetAsync(WBF,    0, (size_t)8 * MB, stream);
  hipMemsetAsync(AGGP,   0, (size_t)8 * 64 * PLANE * 2, stream);
  hipMemsetAsync(Y1P,    0, (size_t)8 * 256 * PLANE * 2, stream);
  hipMemsetAsync(FEATSP, 0, (size_t)8 * 256 * PLANE * 2, stream);
  hipMemsetAsync(AGGTP,  0, (size_t)2 * 64 * PLANE * 2, stream);
  hipMemsetAsync(YTP,    0, (size_t)2 * 256 * PLANE * 2, stream);
  hipMemsetAsync(HFP,    0, (size_t)2 * 256 * PLANE * 2, stream);
  hipMemsetAsync(HBP,    0, (size_t)2 * 64 * PLANE * 2, stream);

  // ---- convert activations + weights ----
  cvt_pad_bf16<<<4096, 256, 0, stream>>>(feat, FEATP, (long)24 * 256);

  size_t wo = 0;
  auto cvtw = [&](const float* src, int Cout, int Cin) -> unsigned short* {
    int R = (Cout + 63) & ~63;
    size_t n = (size_t)9 * R * Cin;
    unsigned short* dst = WBF + wo;
    wo += (n + 63) & ~(size_t)63;
    cvt_wshift<<<256, 256, 0, stream>>>(src, dst, Cout, Cin, R);
    return dst;
  };
  unsigned short* bQW  = cvtw(cf_qw, 64, 256);
  unsigned short* bKW  = cvtw(cf_kw, 64, 256);
  unsigned short* bVW  = cvtw(cf_vw, 64, 256);
  unsigned short* bO1W = cvtw(cf_o1w, 256, 64);
  unsigned short* bO2W = cvtw(cf_o2w, 256, 256);
  unsigned short* bTQW = cvtw(tf_qw, 64, 256);
  unsigned short* bTKW = cvtw(tf_kw, 64, 256);
  unsigned short* bTVW = cvtw(tf_vw, 64, 256);
  unsigned short* bTO1 = cvtw(tf_o1w, 256, 64);
  unsigned short* bTO2 = cvtw(tf_o2w, 256, 256);
  unsigned short* bMW1 = cvtw(mm_w1, 64, 256);
  unsigned short* bMW2 = cvtw(mm_w2, 1, 64);
  unsigned short* bOLW = cvtw(ol_w, 256, 256);

  auto conv = [&](const unsigned short* in, const unsigned short* w, const float* bias,
                  float* out, int Nmaps, int Cin, int Cout) {
    dim3 g(Nmaps * (P / 64), (Cout + 63) / 64);
    conv3x3_wmma<<<g, 128, 0, stream>>>(in, w, bias, out, Cin, Cout);
  };
  auto bn_stats = [&](const float* x, int G, int B, int C) {
    bn_stats_k<<<G * C, 256, 0, stream>>>(x, MEAN, INV, B, C);
  };
  auto bn_relu = [&](const float* x, const float* g, const float* b,
                     unsigned short* outP, int G, int B, int C) {
    bn_relu_bf16_k<<<dim3(G * B * C, P / 256), 256, 0, stream>>>(x, g, b, MEAN, INV,
                                                                 outP, B, C);
  };

  // ================= cf attention (per (i,b) group) =================
  conv(FEATP, bQW, cf_qb, Q, 24, 256, 64);
  conv(FEATP, bKW, cf_kb, Kk, 24, 256, 64);
  conv(FEATP, bVW, cf_vb, V, 24, 256, 64);
  dot_qk<<<24 * 64, 256, 0, stream>>>(Q, Kk, S24);
  softmax64<<<24, 64, 0, stream>>>(S24, SM24);
  const int gst[8] = {0, 4, 6, 9, 12, 14, 18, 21};  // cumsum of CAVS.ravel()
  const int gcn[8] = {4, 2, 3, 3, 2, 4, 3, 3};
  for (int g = 0; g < 8; ++g)
    ctx_agg_k<<<dim3(64, P / 256), 256, 0, stream>>>(V, SM24,
                                                     AGGP + (size_t)g * 64 * PLANE,
                                                     gst[g], 1, gcn[g]);
  conv(AGGP, bO1W, cf_o1b, O1, 8, 64, 256);
  bn_stats(O1, 8, 1, 256);
  bn_relu(O1, cf_g1, cf_b1, Y1P, 8, 1, 256);
  conv(Y1P, bO2W, cf_o2b, O2, 8, 256, 256);
  bn_stats(O2, 8, 1, 256);
  bn_relu(O2, cf_g2, cf_b2, FEATSP, 8, 1, 256);   // FEATSP[g=i*2+b]

  // ================= tf attention on hist = feats[1..3] =================
  const unsigned short* HISTP = FEATSP + (size_t)2 * 256 * PLANE;  // 6 maps, n*2+b
  conv(HISTP, bTQW, tf_qb, QT, 6, 256, 64);
  conv(HISTP, bTKW, tf_kb, KT, 6, 256, 64);
  conv(HISTP, bTVW, tf_vb, VT, 6, 256, 64);
  dot_qk<<<6 * 64, 256, 0, stream>>>(QT, KT, S6);
  softmax64<<<6, 64, 0, stream>>>(S6, SM6);
  for (int b = 0; b < 2; ++b)
    ctx_agg_k<<<dim3(64, P / 256), 256, 0, stream>>>(VT, SM6,
                                                     AGGTP + (size_t)b * 64 * PLANE,
                                                     b, 2, 3);
  conv(AGGTP, bTO1, tf_o1b, O1T, 2, 64, 256);
  bn_stats(O1T, 1, 2, 256);
  bn_relu(O1T, tf_g1, tf_b1, YTP, 1, 2, 256);
  conv(YTP, bTO2, tf_o2b, O2T, 2, 256, 256);
  bn_stats(O2T, 1, 2, 256);
  bn_relu(O2T, tf_g2, tf_b2, HFP, 1, 2, 256);     // hf (padded bf16)

  // ================= mask model =================
  conv(HFP, bMW1, mm_b1, MM1, 2, 256, 64);
  bn_stats(MM1, 1, 2, 64);
  bn_relu(MM1, mm_g1, mm_be1, HBP, 1, 2, 64);
  conv(HBP, bMW2, mm_b2, MM2, 2, 64, 1);
  bn_stats(MM2, 1, 2, 1);
  float* OUT = (float*)d_out;                     // m: [2][P], then mhf: [2][256][P]
  bn_sigmoid_f32_k<<<dim3(2, P / 256), 256, 0, stream>>>(MM2, mm_g2, mm_be2,
                                                         MEAN, INV, OUT, 2, 1);

  // ================= output layer: relu(bn(conv(hf))) * m =================
  conv(HFP, bOLW, ol_b, OLO, 2, 256, 256);
  bn_stats(OLO, 1, 2, 256);
  bn_relu_mask_f32_k<<<dim3(2 * 256, P / 256), 256, 0, stream>>>(
      OLO, ol_g, ol_be, MEAN, INV, OUT, OUT + 2 * (size_t)P, 2, 256);
}
